// GCN_dc_4612794876642
// MI455X (gfx1250) — compile-verified, matrix operand order checked by reading, not run on hardware
//
#include <hip/hip_runtime.h>
#include <hip/hip_bf16.h>
#include <math.h>

// ---------------------------------------------------------------------------
// Types for CDNA5 WMMA (wave32, v_wmma_f32_16x16x32_bf16)
// ---------------------------------------------------------------------------
typedef __attribute__((ext_vector_type(16))) __bf16 v16bf;
typedef __attribute__((ext_vector_type(8)))  __bf16 v8bf;
typedef __attribute__((ext_vector_type(8)))  float  v8f;

#define NODES_PAD 100032   // 64 * 1563, multiple of the 64-row M tile
#define NHID 256

// ---------------------------------------------------------------------------
// Fragment load for 16-bit 16x32 A/B operand (CDNA5 layout):
//   lane l: row/col = l%16 ; half = l/16
//   VGPR v (v<4):  k = half*8 + 2v, 2v+1        -> bytes [half*16, half*16+16)
//   VGPR v (v>=4): k = 16 + half*8 + ...        -> bytes [32+half*16, ...+16)
// Two contiguous 128-bit loads per fragment (global_load_b128 / ds_load_b128).
// ---------------------------------------------------------------------------
__device__ __forceinline__ v16bf load_frag(const __bf16* __restrict__ p, int half) {
    const v8bf* q0 = reinterpret_cast<const v8bf*>(p + (half << 3));
    const v8bf* q1 = reinterpret_cast<const v8bf*>(p + 16 + (half << 3));
    v8bf lo = *q0;
    v8bf hi = *q1;
    return __builtin_shufflevector(lo, hi, 0, 1, 2, 3, 4, 5, 6, 7,
                                           8, 9, 10, 11, 12, 13, 14, 15);
}

// ---------------------------------------------------------------------------
// WMMA GEMM: C[M x 64*gridDim.y] = A[M x K](bf16) * Wt^T, Wt is [Nout x K] bf16
// rows contiguous (ldw == K). The block's 64-row weight group is staged into
// LDS once (64*K*2 bytes dynamic shared) and B fragments come from LDS; A
// fragments are register double-buffered to overlap VMEM latency with WMMA.
// Block = 128 threads = 4 waves; wave w -> rows m0+16w .. +15; 4 N-tiles each.
// M must be a multiple of 64 (node buffers padded to NODES_PAD).
// ---------------------------------------------------------------------------
__global__ void __launch_bounds__(128)
k_gemm_bf16(const __bf16* __restrict__ A, int lda,
            const __bf16* __restrict__ Wt,
            float* __restrict__ C, int ldc, int K) {
    extern __shared__ __bf16 ldsB[];            // [64][K]
    const int lane = threadIdx.x & 31;
    const int wv   = threadIdx.x >> 5;
    const int half = lane >> 4;
    const int lm   = lane & 15;
    const int m0   = blockIdx.x * 64 + wv * 16;
    const int nb   = blockIdx.y * 64;

    // ---- stage the 64 x K weight group into LDS (16B chunks, all threads) ----
    {
        const uint4* src = reinterpret_cast<const uint4*>(Wt + (size_t)nb * K);
        uint4*       dst = reinterpret_cast<uint4*>(ldsB);
        const int nchunks = (64 * K) >> 3;      // 8 bf16 per 16B chunk
        for (int i = threadIdx.x; i < nchunks; i += 128) dst[i] = src[i];
    }
    __syncthreads();

    const __bf16* arow = A + (size_t)(m0 + lm) * lda;
    const __bf16* b0   = ldsB + (size_t)(lm)      * K;
    const __bf16* b1   = ldsB + (size_t)(16 + lm) * K;
    const __bf16* b2   = ldsB + (size_t)(32 + lm) * K;
    const __bf16* b3   = ldsB + (size_t)(48 + lm) * K;

    v8f acc0 = {}, acc1 = {}, acc2 = {}, acc3 = {};

    v16bf a_cur = load_frag(arow, half);        // prime the A pipeline
    for (int kk = 0; kk < K; kk += 32) {
        v16bf a_nxt = a_cur;
        if (kk + 32 < K) {                      // issue next A load early
            __builtin_prefetch(arow + kk + 96, 0, 1);
            a_nxt = load_frag(arow + kk + 32, half);
        }
        v16bf f0 = load_frag(b0 + kk, half);
        v16bf f1 = load_frag(b1 + kk, half);
        v16bf f2 = load_frag(b2 + kk, half);
        v16bf f3 = load_frag(b3 + kk, half);
        acc0 = __builtin_amdgcn_wmma_f32_16x16x32_bf16(false, a_cur, false, f0,
                                                       (short)0, acc0, false, false);
        acc1 = __builtin_amdgcn_wmma_f32_16x16x32_bf16(false, a_cur, false, f1,
                                                       (short)0, acc1, false, false);
        acc2 = __builtin_amdgcn_wmma_f32_16x16x32_bf16(false, a_cur, false, f2,
                                                       (short)0, acc2, false, false);
        acc3 = __builtin_amdgcn_wmma_f32_16x16x32_bf16(false, a_cur, false, f3,
                                                       (short)0, acc3, false, false);
        a_cur = a_nxt;
    }

    // C/D layout: VGPR r: lanes 0-15 -> row m0+r, lanes 16-31 -> row m0+8+r; col = lane%16
    const int rbase = m0 + half * 8;
#pragma unroll
    for (int r = 0; r < 8; ++r) {
        float* crow = C + (size_t)(rbase + r) * ldc + lm;
        crow[nb +  0] = acc0[r];
        crow[nb + 16] = acc1[r];
        crow[nb + 32] = acc2[r];
        crow[nb + 48] = acc3[r];
    }
}

// ---------------------------------------------------------------------------
// Small helper kernels
// ---------------------------------------------------------------------------
__global__ void k_fill_f32(float* p, float v, long n) {
    long i = (long)blockIdx.x * blockDim.x + threadIdx.x;
    if (i < n) p[i] = v;
}

__global__ void k_deg_count(const int* __restrict__ dst, float* __restrict__ deg, int E) {
    int e = blockIdx.x * blockDim.x + threadIdx.x;
    if (e < E) atomicAdd(&deg[dst[e]], 1.0f);
}

__global__ void k_dinv(const float* __restrict__ deg, float* __restrict__ dinv, int n) {
    int i = blockIdx.x * blockDim.x + threadIdx.x;
    if (i < n) dinv[i] = rsqrtf(fmaxf(deg[i], 1.0f));   // deg >= 1 (self loop)
}

// Transpose + bf16-cast + zero-pad weights: W[K x Ncols] f32 -> Wt[Nout x Kpad] bf16
__global__ void k_transpose_w(const float* __restrict__ W, int K, int Ncols,
                              __bf16* __restrict__ Wt, int Kpad, int Nout) {
    long idx = (long)blockIdx.x * blockDim.x + threadIdx.x;
    if (idx >= (long)Nout * Kpad) return;
    int n = (int)(idx / Kpad);
    int k = (int)(idx % Kpad);
    float v = (k < K && n < Ncols) ? W[(size_t)k * Ncols + n] : 0.0f;
    Wt[idx] = (__bf16)v;
}

// Pack x (f32 [N,500]) into A1 bf16 [NODES_PAD,768] cols 0..499; cols 756..767 = 0.
__global__ void k_pack_x(const float* __restrict__ x, __bf16* __restrict__ A1, int N) {
    long idx = (long)blockIdx.x * blockDim.x + threadIdx.x;
    if (idx >= (long)NODES_PAD * 768) return;
    int node = (int)(idx / 768);
    int c    = (int)(idx % 768);
    if (c >= 500 && c < 756) return;                // embed kernel owns these
    int rn = node < N ? node : N - 1;               // clamp pad rows (finite junk ok)
    float v = (c < 500) ? x[(size_t)rn * 500 + c] : 0.0f;
    A1[idx] = (__bf16)v;
}

// s = s_raw @ We + be  (K=16); write bf16 into A1 cols 500..755 and sbf1.
__global__ void k_embed(const float* __restrict__ s_raw, const float* __restrict__ We,
                        const float* __restrict__ be, __bf16* __restrict__ A1,
                        __bf16* __restrict__ sbf, int N) {
    long idx = (long)blockIdx.x * blockDim.x + threadIdx.x;
    if (idx >= (long)NODES_PAD * NHID) return;
    int node = (int)(idx / NHID);
    int c    = (int)(idx % NHID);
    int rn = node < N ? node : N - 1;
    float acc = be[c];
#pragma unroll
    for (int k = 0; k < 16; ++k)
        acc = fmaf(s_raw[(size_t)rn * 16 + k], We[(size_t)k * NHID + c], acc);
    sbf[(size_t)node * NHID + c] = (__bf16)acc;
    A1[(size_t)node * 768 + 500 + c] = (__bf16)acc;
}

// Edge scatter: agg[dst] += t[src] * dinv[src]*dinv[dst]   (one block per edge)
__global__ void __launch_bounds__(256)
k_aggregate(const float* __restrict__ t, const int* __restrict__ src,
            const int* __restrict__ dst, const float* __restrict__ dinv,
            float* __restrict__ agg, int E) {
    int e = blockIdx.x;
    if (e >= E) return;
    int s = src[e], d = dst[e];
    float nm = dinv[s] * dinv[d];
    int c = threadIdx.x;
    atomicAdd(&agg[(size_t)d * NHID + c], t[(size_t)s * NHID + c] * nm);
}

// finalize GCN: v = act(agg + t*dinv^2 + b); write bf16 to up to two slots
// ACT 0 = relu, 1 = tanh
template <int ACT>
__global__ void k_finalize_gcn(const float* __restrict__ agg, const float* __restrict__ t,
                               const float* __restrict__ dinv, const float* __restrict__ b,
                               __bf16* __restrict__ o0, int p0, int c0,
                               __bf16* __restrict__ o1, int p1, int c1) {
    long idx = (long)blockIdx.x * blockDim.x + threadIdx.x;
    if (idx >= (long)NODES_PAD * NHID) return;
    int node = (int)(idx / NHID);
    int c    = (int)(idx % NHID);
    float di = dinv[node];
    float v  = agg[idx] + t[idx] * di * di + b[c];
    v = (ACT == 0) ? fmaxf(v, 0.0f) : tanhf(v);
    __bf16 bv = (__bf16)v;
    if (o0) o0[(size_t)node * p0 + c0 + c] = bv;
    if (o1) o1[(size_t)node * p1 + c1 + c] = bv;
}

// dense finalize: hp = relu(t + bp) -> bf16
__global__ void k_finalize_dense(const float* __restrict__ t, const float* __restrict__ b,
                                 __bf16* __restrict__ out) {
    long idx = (long)blockIdx.x * blockDim.x + threadIdx.x;
    if (idx >= (long)NODES_PAD * NHID) return;
    int c = (int)(idx % NHID);
    out[idx] = (__bf16)fmaxf(t[idx] + b[c], 0.0f);
}

// log_softmax over 16 classes; logits stored padded [NODES_PAD,64]
__global__ void k_logsoftmax(const float* __restrict__ lg, const float* __restrict__ bc,
                             float* __restrict__ out, int N) {
    int node = blockIdx.x * blockDim.x + threadIdx.x;
    if (node >= N) return;
    const float* lp = lg + (size_t)node * 64;
    float v[16], mx = -3.4e38f;
#pragma unroll
    for (int j = 0; j < 16; ++j) { v[j] = lp[j] + bc[j]; mx = fmaxf(mx, v[j]); }
    float s = 0.0f;
#pragma unroll
    for (int j = 0; j < 16; ++j) s += __expf(v[j] - mx);
    float lse = mx + __logf(s);
#pragma unroll
    for (int j = 0; j < 16; ++j) out[(size_t)node * 16 + j] = v[j] - lse;
}

// ---------------------------------------------------------------------------
// Host launcher
// ---------------------------------------------------------------------------
extern "C" void kernel_launch(void* const* d_in, const int* in_sizes, int n_in,
                              void* d_out, int out_size, void* d_ws, size_t ws_size,
                              hipStream_t stream) {
    const float* x     = (const float*)d_in[0];
    const float* s_raw = (const float*)d_in[1];
    const float* We    = (const float*)d_in[2];
    const float* be    = (const float*)d_in[3];
    const float* W1    = (const float*)d_in[4];
    const float* b1    = (const float*)d_in[5];
    const float* Wg1   = (const float*)d_in[6];
    const float* bg1   = (const float*)d_in[7];
    const float* W2    = (const float*)d_in[8];
    const float* b2    = (const float*)d_in[9];
    const float* Wg2   = (const float*)d_in[10];
    const float* bg2   = (const float*)d_in[11];
    const float* Wp    = (const float*)d_in[12];
    const float* bp    = (const float*)d_in[13];
    const float* Wc    = (const float*)d_in[14];
    const float* bc    = (const float*)d_in[15];
    const int*   srcI  = (const int*)d_in[16];
    const int*   dstI  = (const int*)d_in[17];

    const int N = in_sizes[1] / 16;      // 100000
    const int E = in_sizes[16];          // 1600000
    const int NP = NODES_PAD;

    // ---- carve workspace (bump allocator, 256B aligned) ----
    uint8_t* base = (uint8_t*)d_ws;
    size_t off = 0;
    auto alloc = [&](size_t bytes) -> void* {
        void* p = base + off;
        off += (bytes + 255) & ~(size_t)255;
        return p;
    };
    float*  deg   = (float*)alloc((size_t)NP * 4);
    float*  dinv  = (float*)alloc((size_t)NP * 4);
    __bf16* W1t   = (__bf16*)alloc((size_t)256 * 768 * 2);
    __bf16* Wg1t  = (__bf16*)alloc((size_t)256 * 256 * 2);
    __bf16* W2t   = (__bf16*)alloc((size_t)256 * 512 * 2);
    __bf16* Wg2t  = (__bf16*)alloc((size_t)256 * 256 * 2);
    __bf16* Wpt   = (__bf16*)alloc((size_t)256 * 512 * 2);
    __bf16* Wct   = (__bf16*)alloc((size_t)64  * 256 * 2);
    __bf16* A1    = (__bf16*)alloc((size_t)NP * 768 * 2);
    __bf16* sbf1  = (__bf16*)alloc((size_t)NP * 256 * 2);
    __bf16* sbf2  = (__bf16*)alloc((size_t)NP * 256 * 2);
    __bf16* cat1  = (__bf16*)alloc((size_t)NP * 512 * 2);
    __bf16* cat2  = (__bf16*)alloc((size_t)NP * 512 * 2);
    float*  t     = (float*)alloc((size_t)NP * 256 * 4);
    float*  agg   = (float*)alloc((size_t)NP * 256 * 4);
    if (off > ws_size) return;   // workspace too small; bail deterministically

    const int TB = 256;
    auto blocks = [](long n, int tb) { return (unsigned)((n + tb - 1) / tb); };
    auto shmem  = [](int K) { return (size_t)64 * K * 2; };   // B-group staging bytes
    const long NPH = (long)NP * NHID;

    // ---- weight transposes (bf16, zero-padded) ----
    k_transpose_w<<<blocks((long)256 * 768, TB), TB, 0, stream>>>(W1, 756, 256, W1t, 768, 256);
    k_transpose_w<<<blocks((long)256 * 256, TB), TB, 0, stream>>>(Wg1, 256, 256, Wg1t, 256, 256);
    k_transpose_w<<<blocks((long)256 * 512, TB), TB, 0, stream>>>(W2, 512, 256, W2t, 512, 256);
    k_transpose_w<<<blocks((long)256 * 256, TB), TB, 0, stream>>>(Wg2, 256, 256, Wg2t, 256, 256);
    k_transpose_w<<<blocks((long)256 * 512, TB), TB, 0, stream>>>(Wp, 512, 256, Wpt, 512, 256);
    k_transpose_w<<<blocks((long)64 * 256, TB), TB, 0, stream>>>(Wc, 256, 16, Wct, 256, 64);

    // ---- edge norms ----
    k_fill_f32<<<blocks(NP, TB), TB, 0, stream>>>(deg, 1.0f, NP);          // self-loop
    k_deg_count<<<blocks(E, TB), TB, 0, stream>>>(dstI, deg, E);
    k_dinv<<<blocks(NP, TB), TB, 0, stream>>>(deg, dinv, NP);

    // ---- inputs: pack x, embed s ----
    k_pack_x<<<blocks((long)NP * 768, TB), TB, 0, stream>>>(x, A1, N);
    k_embed<<<blocks(NPH, TB), TB, 0, stream>>>(s_raw, We, be, A1, sbf1, N);

    dim3 gemmGrid(NP / 64, 4), gemmBlk(128);

    // ---- GCN layer 1 (h): relu(gcn([x|s] @ W1)) -> cat1[:,0:256] ----
    k_gemm_bf16<<<gemmGrid, gemmBlk, shmem(768), stream>>>(A1, 768, W1t, t, 256, 768);
    k_fill_f32<<<blocks(NPH, TB), TB, 0, stream>>>(agg, 0.0f, NPH);
    k_aggregate<<<E, 256, 0, stream>>>(t, srcI, dstI, dinv, agg, E);
    k_finalize_gcn<0><<<blocks(NPH, TB), TB, 0, stream>>>(agg, t, dinv, b1,
                                                          cat1, 512, 0, nullptr, 0, 0);
    // ---- GCN layer 1 (s): tanh(gcn(s @ Wg1)) -> sbf2, cat1[:,256:512] ----
    k_gemm_bf16<<<gemmGrid, gemmBlk, shmem(256), stream>>>(sbf1, 256, Wg1t, t, 256, 256);
    k_fill_f32<<<blocks(NPH, TB), TB, 0, stream>>>(agg, 0.0f, NPH);
    k_aggregate<<<E, 256, 0, stream>>>(t, srcI, dstI, dinv, agg, E);
    k_finalize_gcn<1><<<blocks(NPH, TB), TB, 0, stream>>>(agg, t, dinv, bg1,
                                                          sbf2, 256, 0, cat1, 512, 256);
    // ---- GCN layer 2 (h): relu(gcn(cat1 @ W2)) -> cat2[:,0:256] ----
    k_gemm_bf16<<<gemmGrid, gemmBlk, shmem(512), stream>>>(cat1, 512, W2t, t, 256, 512);
    k_fill_f32<<<blocks(NPH, TB), TB, 0, stream>>>(agg, 0.0f, NPH);
    k_aggregate<<<E, 256, 0, stream>>>(t, srcI, dstI, dinv, agg, E);
    k_finalize_gcn<0><<<blocks(NPH, TB), TB, 0, stream>>>(agg, t, dinv, b2,
                                                          cat2, 512, 0, nullptr, 0, 0);
    // ---- GCN layer 2 (s): tanh(gcn(sbf2 @ Wg2)) -> cat2[:,256:512] ----
    k_gemm_bf16<<<gemmGrid, gemmBlk, shmem(256), stream>>>(sbf2, 256, Wg2t, t, 256, 256);
    k_fill_f32<<<blocks(NPH, TB), TB, 0, stream>>>(agg, 0.0f, NPH);
    k_aggregate<<<E, 256, 0, stream>>>(t, srcI, dstI, dinv, agg, E);
    k_finalize_gcn<1><<<blocks(NPH, TB), TB, 0, stream>>>(agg, t, dinv, bg2,
                                                          nullptr, 0, 0, cat2, 512, 256);
    // ---- dense projection: hp = relu(cat2 @ Wp + bp) -> sbf1 (bf16) ----
    k_gemm_bf16<<<gemmGrid, gemmBlk, shmem(512), stream>>>(cat2, 512, Wpt, t, 256, 512);
    k_finalize_dense<<<blocks(NPH, TB), TB, 0, stream>>>(t, bp, sbf1);

    // ---- classifier: logits = hp @ Wc (padded to 64 cols) ----
    dim3 clsGrid(NP / 64, 1);
    k_gemm_bf16<<<clsGrid, gemmBlk, shmem(256), stream>>>(sbf1, 256, Wct, t, 64, 256);

    // ---- log_softmax -> d_out ----
    k_logsoftmax<<<blocks(N, TB), TB, 0, stream>>>(t, bc, (float*)d_out, N);
}
